// EmbeddingBasedNetwork_64029372449203
// MI455X (gfx1250) — compile-verified, matrix operand order checked by reading, not run on hardware
//
#include <hip/hip_runtime.h>
#include <hip/hip_bf16.h>
#include <cstdint>

typedef _Float16 f16;
typedef _Float16 v16h __attribute__((ext_vector_type(16)));
typedef float    v8f  __attribute__((ext_vector_type(8)));

#define USE_ASYNC 1

#define B_ROWS 32768
#define NZ 9
#define NC 52
#define ED 52      // embedding dim
#define ZD 52      // zone encoded dim
#define FD 468     // fusion dim = 9*52
#define FP 480     // fusion dim padded to K-chunk multiple (32)
#define HD 52      // hidden dim
#define HP 64      // hidden padded
#define AD 3157    // num actions
#define AP 3168    // actions padded to 16

// -------- workspace layout (bytes) --------
#define WS_WZ   0
#define WS_WH   8192
#define WS_WO   (8192 + 61440)
#define WS_HID  (8192 + 61440 + 405504)

// Load one 16-half A/B fragment run for v_wmma_*_16x16x32_f16.
// Per ISA 7.12.2 (16-bit A 16x32): lane L holds row M=L%16; halves 0..7 are
// K = koff..koff+7 and halves 8..15 are K = koff+16..koff+23, koff = (L<16)?0:8.
// Caller passes p = &row[kbase + koff].
__device__ __forceinline__ v16h load_frag(const f16* p) {
  v16h r;
#pragma unroll
  for (int i = 0; i < 8; ++i) { r[i] = p[i]; r[i + 8] = p[i + 16]; }
  return r;
}

// ---------------- kernel 0: pack / convert weights to f16 -------------------
__global__ void __launch_bounds__(256)
pack_weights(const float* __restrict__ Wz, const float* __restrict__ Wh,
             const float* __restrict__ Wo,
             f16* __restrict__ wz_h, f16* __restrict__ wh_h, f16* __restrict__ wo_h) {
  int idx = blockIdx.x * 256 + threadIdx.x;
  if (idx < 64 * 64) {                    // W_zone [E=52][Z=52] -> [n=64][k=64]
    int n = idx >> 6, k = idx & 63;
    float v = (n < ZD && k < ED) ? Wz[k * ZD + n] : 0.f;
    wz_h[idx] = (f16)v;
    return;
  }
  int i2 = idx - 64 * 64;
  if (i2 < 64 * FP) {                     // W_hidden [468][52] -> [n=64][k=480]
    int n = i2 / FP, k = i2 % FP;
    float v = (n < HD && k < FD) ? Wh[k * HD + n] : 0.f;
    wh_h[i2] = (f16)v;
    return;
  }
  int i3 = i2 - 64 * FP;
  if (i3 < AP * 64) {                     // W_out [52][3157] -> [n=3168][k=64]
    int n = i3 >> 6, k = i3 & 63;
    float v = (n < AD && k < HD) ? Wo[(long)k * AD + n] : 0.f;
    wo_h[i3] = (f16)v;
  }
}

// ------- kernel 1: masked max + zone encode + hidden, fused per 16 rows -----
__global__ void __launch_bounds__(256)
fused_hidden(const int* __restrict__ zone_masks, const float* __restrict__ emb,
             const float* __restrict__ b_zone, const float* __restrict__ b_hidden,
             const f16* __restrict__ wz_h, const f16* __restrict__ wh_h,
             f16* __restrict__ hid_h) {
  __shared__ float embL[NC][64];          // 13312 B, dim-padded with zeros
  __shared__ f16  zmax[NZ][16][64];       // 18432 B, 9 WMMA A-tiles
  __shared__ f16  fus[16][FP];            // 15360 B, fusion matrix (A of hidden)

  const int tid  = threadIdx.x;
  const int lane = tid & 31;
  const int wave = tid >> 5;
  const int row0 = blockIdx.x * 16;

  // stage embedding [52][64] (zero pad dims 52..63)
  for (int i = tid; i < NC * 64; i += 256) {
    int c = i >> 6, d = i & 63;
    embL[c][d] = (d < ED) ? emb[c * ED + d] : 0.f;
  }
  // zero fusion pad columns 468..479
  for (int i = tid; i < 16 * (FP - FD); i += 256) {
    fus[i / (FP - FD)][FD + i % (FP - FD)] = (f16)0.f;
  }
  __syncthreads();

  // ---- phase A: masked max over cards (2 rows per wave) ----
#pragma unroll
  for (int rr = 0; rr < 2; ++rr) {
    const int rl = wave * 2 + rr;
    const int rg = row0 + rl;
    for (int z = 0; z < NZ; ++z) {
      const int* mrow = zone_masks + ((long)rg * NZ + z) * NC;
      bool p0 = mrow[lane] > 0;
      bool p1 = (lane < NC - 32) ? (mrow[lane + 32] > 0) : false;
      uint32_t m0 = __builtin_amdgcn_ballot_w32(p0);
      uint32_t m1 = __builtin_amdgcn_ballot_w32(p1);
      uint64_t m  = (uint64_t)m0 | ((uint64_t)m1 << 32);
      float a0 = -INFINITY, a1 = -INFINITY;
      uint64_t mm = m;
      while (mm) {                         // iterate only over present cards
        int c = __builtin_ctzll(mm);
        mm &= mm - 1;
        a0 = fmaxf(a0, embL[c][lane]);
        a1 = fmaxf(a1, embL[c][lane + 32]);
      }
      if (m == 0) { a0 = 0.f; a1 = 0.f; }  // empty zone -> zeros
      zmax[z][rl][lane]      = (f16)a0;
      zmax[z][rl][lane + 32] = (lane < ED - 32) ? (f16)a1 : (f16)0.f;
    }
  }
  __syncthreads();

  const int sub  = lane & 15;   // M (for A) or N (for B/C) sub-index
  const int hi   = lane >> 4;
  const int koff = hi ? 8 : 0;

  // ---- phase B: zone encode, 9 M-tiles x 4 N-tiles, WMMA f16 ----
  for (int j = wave; j < NZ * 4; j += 8) {
    const int z = j >> 2, nt = j & 3;
    v8f acc = {};
#pragma unroll
    for (int q = 0; q < 2; ++q) {          // K = 64 (52 real + zero pad)
      v16h a = load_frag(&zmax[z][sub][q * 32 + koff]);
      v16h b = load_frag(wz_h + (nt * 16 + sub) * 64 + q * 32 + koff);
      acc = __builtin_amdgcn_wmma_f32_16x16x32_f16(false, a, false, b,
                                                   (short)0, acc, false, false);
    }
    const int n = nt * 16 + sub;
    if (n < ZD) {
      const float bz = b_zone[n];
#pragma unroll
      for (int i = 0; i < 8; ++i) {
        const int M = i + hi * 8;
        float v = acc[i] + bz;
        fus[M][z * ZD + n] = (f16)(v > 0.f ? v : 0.f);   // relu
      }
    }
  }
  __syncthreads();

  // ---- phase C: hidden layer, M=16, K=480, N=64 (4 N-tiles on waves 0..3) ----
  if (wave < 4) {
    const int nt = wave;
    v8f acc = {};
#pragma unroll
    for (int q = 0; q < FP / 32; ++q) {
      v16h a = load_frag(&fus[sub][q * 32 + koff]);
      v16h b = load_frag(wh_h + (nt * 16 + sub) * FP + q * 32 + koff);
      acc = __builtin_amdgcn_wmma_f32_16x16x32_f16(false, a, false, b,
                                                   (short)0, acc, false, false);
    }
    const int n = nt * 16 + sub;
    const float bh = (n < HD) ? b_hidden[n] : 0.f;
#pragma unroll
    for (int i = 0; i < 8; ++i) {
      const int M = i + hi * 8;
      float v = acc[i] + bh;
      v = (n < HD) ? (v > 0.f ? v : 0.f) : 0.f;          // pad cols stay 0
      hid_h[(long)(row0 + M) * HP + n] = (f16)v;
    }
  }
}

// ------------- kernel 2: q = hidden @ W_out + b_out (the big one) -----------
// 64 batch rows per workgroup: A panel async-staged in LDS once, B fragments
// held in registers across 4 M-tiles (4x less W_out re-read than 16-row WGs).
__global__ void __launch_bounds__(256)
qvalues(const f16* __restrict__ hid_h, const f16* __restrict__ wo_h,
        const float* __restrict__ b_out, float* __restrict__ out) {
  __shared__ f16 aL[64][64];               // 8 KB shared A panel (64 rows)

  const int tid  = threadIdx.x;
  const int lane = tid & 31;
  const int wave = tid >> 5;
  const int row0 = blockIdx.x * 64;

  {
    const f16* g0 = hid_h + (size_t)row0 * 64 + tid * 8;   // 16 B per thread
#if USE_ASYNC
    // LDS byte offset (wave-relative) via explicit AS(3) cast.
    uint32_t lds0 =
        (uint32_t)(uintptr_t)(__attribute__((address_space(3))) f16*)&aL[0][0]
        + (uint32_t)tid * 16u;
    asm volatile("global_load_async_to_lds_b128 %0, %1, off"
                 :: "v"(lds0), "v"(g0) : "memory");
    asm volatile("global_load_async_to_lds_b128 %0, %1, off"
                 :: "v"(lds0 + 4096u), "v"(g0 + 2048) : "memory");
    asm volatile("s_wait_asynccnt 0x0" ::: "memory");
#else
    f16* l0 = &aL[0][0] + tid * 8;
#pragma unroll
    for (int i = 0; i < 8; ++i) l0[i] = g0[i];
#pragma unroll
    for (int i = 0; i < 8; ++i) l0[2048 + i] = g0[2048 + i];
#endif
  }
  __syncthreads();

  const int ntile = blockIdx.y * 8 + wave;
  if (ntile >= AP / 16) return;

  const int sub  = lane & 15;
  const int hi   = lane >> 4;
  const int koff = hi ? 8 : 0;

  // B fragments: loaded once, reused across all 4 M-tiles
  const f16* bcol = wo_h + (size_t)(ntile * 16 + sub) * 64 + koff;
  const v16h b0 = load_frag(bcol);
  const v16h b1 = load_frag(bcol + 32);

  const int n = ntile * 16 + sub;
  const float bo = (n < AD) ? b_out[n] : 0.f;

#pragma unroll
  for (int mt = 0; mt < 4; ++mt) {
    const f16* arow = &aL[mt * 16 + sub][koff];
    v16h a0 = load_frag(arow);
    v16h a1 = load_frag(arow + 32);
    v8f acc = {};
    acc = __builtin_amdgcn_wmma_f32_16x16x32_f16(false, a0, false, b0,
                                                 (short)0, acc, false, false);
    acc = __builtin_amdgcn_wmma_f32_16x16x32_f16(false, a1, false, b1,
                                                 (short)0, acc, false, false);
    if (n < AD) {
#pragma unroll
      for (int i = 0; i < 8; ++i) {
        const int M = mt * 16 + i + hi * 8;
        out[(size_t)(row0 + M) * AD + n] = acc[i] + bo;
      }
    }
  }
}

extern "C" void kernel_launch(void* const* d_in, const int* in_sizes, int n_in,
                              void* d_out, int out_size, void* d_ws, size_t ws_size,
                              hipStream_t stream) {
  const int*   zone_masks = (const int*)  d_in[0];
  const float* emb        = (const float*)d_in[1];
  const float* Wz         = (const float*)d_in[2];
  const float* bz         = (const float*)d_in[3];
  const float* Wh         = (const float*)d_in[4];
  const float* bh         = (const float*)d_in[5];
  const float* Wo         = (const float*)d_in[6];
  const float* bo         = (const float*)d_in[7];
  float* out = (float*)d_out;

  char* ws = (char*)d_ws;
  f16* wz_h  = (f16*)(ws + WS_WZ);
  f16* wh_h  = (f16*)(ws + WS_WH);
  f16* wo_h  = (f16*)(ws + WS_WO);
  f16* hid_h = (f16*)(ws + WS_HID);

  // total pack elements = 64*64 + 64*480 + 3168*64 = 237568 = 928 * 256
  pack_weights<<<928, 256, 0, stream>>>(Wz, Wh, Wo, wz_h, wh_h, wo_h);
  fused_hidden<<<B_ROWS / 16, 256, 0, stream>>>(zone_masks, emb, bz, bh,
                                                wz_h, wh_h, hid_h);
  dim3 g2(B_ROWS / 64, (AP / 16 + 7) / 8);   // 512 x 25
  qvalues<<<g2, 256, 0, stream>>>(hid_h, wo_h, bo, out);
}
